// Attention_54589034332289
// MI455X (gfx1250) — matrix operand order repaired and checked by reading, not compiled, not run on hardware
//
#include <hip/hip_runtime.h>

#define EMBED 768
#define NH 12
#define HD 64
#define BATCH 16
#define SEQ 1024

typedef float    v8f  __attribute__((ext_vector_type(8)));
typedef float    v4f  __attribute__((ext_vector_type(4)));
typedef _Float16 v16h __attribute__((ext_vector_type(16)));
typedef _Float16 v8h  __attribute__((ext_vector_type(8)));

__device__ __forceinline__ v16h cat8(v8h lo, v8h hi) {
  return __builtin_shufflevector(lo, hi, 0,1,2,3,4,5,6,7,8,9,10,11,12,13,14,15);
}

__device__ __forceinline__ v8f wmma_f16(v16h a, v16h b, v8f c) {
  // D = A(16x32 f16) x B(32x16 f16) + C(16x16 f32)
  return __builtin_amdgcn_wmma_f32_16x16x32_f16(false, a, false, b, (short)0, c, false, false);
}

// ---------------------------------------------------------------------------
// Kernel 1: QKV projection. out = x @ w_qkv + b_qkv, scattered to Q/K/V f16
// scratch in [B, H, N, D] layout. x, w are f32 in global; converted to f16
// while staging into LDS. Block tile 128x128, 8 waves of 64x32 each.
// B tile staged transposed with packed ds_store_b32 (full LDS bank width).
// ---------------------------------------------------------------------------
__global__ __launch_bounds__(256) void qkv_gemm_kernel(
    const float* __restrict__ x, const float* __restrict__ w,
    const float* __restrict__ bias,
    _Float16* __restrict__ Q, _Float16* __restrict__ K, _Float16* __restrict__ V)
{
  constexpr int LDA = 40, LDB = 40, NOUT = 3 * EMBED;
  __shared__ _Float16 As[128 * LDA];   // [m within tile][k]  (row-major)
  __shared__ _Float16 Bs[128 * LDB];   // [n within tile][k]  (transposed w)

  const int t    = threadIdx.x;
  const int lane = t & 31, wave = t >> 5;
  const int lg   = lane >> 4, lr = lane & 15;
  const int wm   = wave >> 2, wn = wave & 3;      // waves as 2 (M) x 4 (N)
  const int mBase = blockIdx.y * 128;
  const int nBase = blockIdx.x * 128;

  v8f acc[4][2];
#pragma unroll
  for (int i = 0; i < 4; ++i)
#pragma unroll
    for (int j = 0; j < 2; ++j) acc[i][j] = (v8f){0,0,0,0,0,0,0,0};

  const int arow = t >> 1, aseg = t & 1;          // A staging: 2 thr/row
  const int bkp  = t >> 4, bns  = t & 15;         // B staging: k-pair x 8-n

  for (int kb = 0; kb < EMBED; kb += 32) {
    __syncthreads();
    // ---- stage A tile: x f32 -> f16 into As[m][k] ----
    {
      const float* xp = x + (size_t)(mBase + arow) * EMBED + kb + aseg * 16;
      v4f f0 = *(const v4f*)(xp + 0);
      v4f f1 = *(const v4f*)(xp + 4);
      v4f f2 = *(const v4f*)(xp + 8);
      v4f f3 = *(const v4f*)(xp + 12);
      v8h c0, c1;
#pragma unroll
      for (int i = 0; i < 4; ++i) {
        c0[i]     = (_Float16)f0[i];
        c0[i + 4] = (_Float16)f1[i];
        c1[i]     = (_Float16)f2[i];
        c1[i + 4] = (_Float16)f3[i];
      }
      *(v8h*)&As[arow * LDA + aseg * 16]     = c0;
      *(v8h*)&As[arow * LDA + aseg * 16 + 8] = c1;
    }
    // ---- stage B tile transposed: rows (k, k+1) for 8 n's, packed b32 ----
    {
      const float* w0 = w + (size_t)(kb + 2 * bkp) * NOUT + nBase + bns * 8;
      const float* w1 = w0 + NOUT;
      v4f r00 = *(const v4f*)(w0);
      v4f r01 = *(const v4f*)(w0 + 4);
      v4f r10 = *(const v4f*)(w1);
      v4f r11 = *(const v4f*)(w1 + 4);
      float f0[8], f1[8];
#pragma unroll
      for (int i = 0; i < 4; ++i) {
        f0[i] = r00[i]; f0[i + 4] = r01[i];
        f1[i] = r10[i]; f1[i + 4] = r11[i];
      }
#pragma unroll
      for (int i = 0; i < 8; ++i) {
        union { _Float16 h[2]; unsigned u; } pk;
        pk.h[0] = (_Float16)f0[i];       // k   of row n
        pk.h[1] = (_Float16)f1[i];       // k+1 of row n
        *(unsigned*)&Bs[(bns * 8 + i) * LDB + 2 * bkp] = pk.u;
      }
    }
    __syncthreads();

    // ---- fragments + 8 WMMA ----
    v16h af[4], bf[2];
#pragma unroll
    for (int mt = 0; mt < 4; ++mt) {
      const _Float16* ap = &As[(wm * 64 + mt * 16 + lr) * LDA + lg * 8];
      af[mt] = cat8(*(const v8h*)ap, *(const v8h*)(ap + 16));
    }
#pragma unroll
    for (int nt = 0; nt < 2; ++nt) {
      const _Float16* bp = &Bs[(wn * 32 + nt * 16 + lr) * LDB + lg * 8];
      bf[nt] = cat8(*(const v8h*)bp, *(const v8h*)(bp + 16));
    }
#pragma unroll
    for (int mt = 0; mt < 4; ++mt)
#pragma unroll
      for (int nt = 0; nt < 2; ++nt)
        acc[mt][nt] = wmma_f16(af[mt], bf[nt], acc[mt][nt]);
  }

  // ---- epilogue: bias + scatter to Q/K/V [B,H,N,D] f16 ----
#pragma unroll
  for (int nt = 0; nt < 2; ++nt) {
    const int col = nBase + wn * 32 + nt * 16 + lr;     // [0, 2304)
    const int s = col / EMBED, r = col % EMBED;
    const int h = r >> 6, d = r & 63;
    _Float16* dst = (s == 0) ? Q : (s == 1) ? K : V;
    const float bv = bias[col];
#pragma unroll
    for (int mt = 0; mt < 4; ++mt) {
#pragma unroll
      for (int j = 0; j < 8; ++j) {
        const int row = mBase + wm * 64 + mt * 16 + j + 8 * lg;
        const int bb = row >> 10, n = row & 1023;
        dst[(((size_t)(bb * NH + h)) * SEQ + n) * HD + d] =
            (_Float16)(acc[mt][nt][j] + bv);
      }
    }
  }
}

// ---------------------------------------------------------------------------
// Kernel 2: flash attention. grid = (SEQ/128, NH, BATCH), 8 waves/block,
// each wave owns 16 query rows. Per 32-key step: 4 wmma scores + online
// softmax + 4 wmma for O += P*V. Output f16 in [B, N, E] layout.
// ---------------------------------------------------------------------------
__global__ __launch_bounds__(256) void attn_kernel(
    const _Float16* __restrict__ Qh, const _Float16* __restrict__ Kh,
    const _Float16* __restrict__ Vh, _Float16* __restrict__ AO)
{
  constexpr int LV = 40, LP = 40;
  __shared__ _Float16 Vs[64 * LV];           // transposed V tile [d][m']
  __shared__ _Float16 Ps[8 * 16 * LP];       // per-wave P tiles  [n][m']

  const int t    = threadIdx.x;
  const int lane = t & 31, wave = t >> 5;
  const int lg   = lane >> 4, lr = lane & 15;
  const int h = blockIdx.y, b = blockIdx.z;
  const int r0 = blockIdx.x * 128 + wave * 16;
  const size_t headOff = ((size_t)(b * NH + h)) * SEQ * HD;

  // Q fragments (A-layout, 16x64 over two K=32 halves), pre-scaled by D^-0.5
  v16h qf[2];
  {
    const _Float16* qp = Qh + headOff + (size_t)(r0 + lr) * HD;
#pragma unroll
    for (int c = 0; c < 2; ++c) {
      const _Float16* p = qp + c * 32 + lg * 8;
      v16h qv = cat8(*(const v8h*)p, *(const v8h*)(p + 16));
      qf[c] = qv * (_Float16)0.125f;         // 1/sqrt(64)
    }
  }

  v8f o[4];
#pragma unroll
  for (int i = 0; i < 4; ++i) o[i] = (v8f){0,0,0,0,0,0,0,0};
  float rm[8], rl[8];
#pragma unroll
  for (int j = 0; j < 8; ++j) { rm[j] = -3.0e38f; rl[j] = 0.0f; }

  _Float16* Pw = &Ps[wave * 16 * LP];

  for (int mb = 0; mb < SEQ; mb += 32) {
    __syncthreads();
    // ---- cooperatively stage V tile transposed into LDS ----
    {
      const int mm = t >> 3, d0 = (t & 7) * 8;
      const _Float16* vp = Vh + headOff + (size_t)(mb + mm) * HD + d0;
      v8h vv = *(const v8h*)vp;
#pragma unroll
      for (int i = 0; i < 8; ++i) Vs[(d0 + i) * LV + mm] = vv[i];
    }
    __syncthreads();

    // ---- scores S(16x32) = Q x K^T, K B-fragments straight from global ----
    v8f s0 = (v8f){0,0,0,0,0,0,0,0};
    v8f s1 = (v8f){0,0,0,0,0,0,0,0};
#pragma unroll
    for (int c = 0; c < 2; ++c) {
      const _Float16* kp0 = Kh + headOff + (size_t)(mb + lr) * HD + c * 32 + lg * 8;
      const _Float16* kp1 = kp0 + 16 * HD;
      v16h kb0 = cat8(*(const v8h*)kp0, *(const v8h*)(kp0 + 16));
      v16h kb1 = cat8(*(const v8h*)kp1, *(const v8h*)(kp1 + 16));
      s0 = wmma_f16(qf[c], kb0, s0);
      s1 = wmma_f16(qf[c], kb1, s1);
    }

    // ---- online softmax (row = VGPR index, cols = 16-lane group) ----
#pragma unroll
    for (int j = 0; j < 8; ++j) {
      float a0 = s0[j], a1 = s1[j];
      float mcur = fmaxf(a0, a1);
#pragma unroll
      for (int off = 1; off < 16; off <<= 1)
        mcur = fmaxf(mcur, __shfl_xor(mcur, off, 16));
      const float mnew = fmaxf(rm[j], mcur);
      const float corr = __expf(rm[j] - mnew);
      const float p0 = __expf(a0 - mnew);
      const float p1 = __expf(a1 - mnew);
      float rs = p0 + p1;
#pragma unroll
      for (int off = 1; off < 16; off <<= 1)
        rs += __shfl_xor(rs, off, 16);
      rl[j] = rl[j] * corr + rs;
      rm[j] = mnew;
      o[0][j] *= corr; o[1][j] *= corr; o[2][j] *= corr; o[3][j] *= corr;
      const int prow = j + 8 * lg;
      Pw[prow * LP + lr]      = (_Float16)p0;
      Pw[prow * LP + 16 + lr] = (_Float16)p1;
    }

    asm volatile("s_wait_dscnt 0" ::: "memory");

    // ---- repack P into A-layout fragment ----
    const _Float16* pp = &Pw[lr * LP + lg * 8];
    v16h pf = cat8(*(const v8h*)pp, *(const v8h*)(pp + 16));

    // ---- O(16x64) += P(16x32) x V(32x64) ----
#pragma unroll
    for (int dt = 0; dt < 4; ++dt) {
      const _Float16* vpp = &Vs[(dt * 16 + lr) * LV + lg * 8];
      v16h vf = cat8(*(const v8h*)vpp, *(const v8h*)(vpp + 16));
      o[dt] = wmma_f16(pf, vf, o[dt]);
    }

    // prefetch next K tile rows (emits global_prefetch_b8)
    if (mb + 32 < SEQ)
      __builtin_prefetch(Kh + headOff + (size_t)(mb + 32 + lr) * HD, 0, 1);
  }

  // ---- epilogue: normalize and store f16 to [B, N, E] ----
  float inv[8];
#pragma unroll
  for (int j = 0; j < 8; ++j) inv[j] = 1.0f / rl[j];
#pragma unroll
  for (int dt = 0; dt < 4; ++dt) {
#pragma unroll
    for (int j = 0; j < 8; ++j) {
      const int n = r0 + j + 8 * lg;
      const int d = dt * 16 + lr;
      AO[((size_t)b * SEQ + n) * EMBED + h * HD + d] =
          (_Float16)(o[dt][j] * inv[j]);
    }
  }
}

// ---------------------------------------------------------------------------
// Kernel 3: output projection. out(f32) = AO(f16) @ w_proj(f32->f16) + b_proj.
// A tile is a pure f16 copy -> use async global->LDS DMA (ASYNCcnt path),
// no VGPR round-trip. B tile converted + transposed with packed b32 stores.
// ---------------------------------------------------------------------------
__global__ __launch_bounds__(256) void proj_gemm_kernel(
    const _Float16* __restrict__ A, const float* __restrict__ w,
    const float* __restrict__ bias, float* __restrict__ out)
{
  constexpr int LDA = 40, LDB = 40;
  __shared__ _Float16 As[128 * LDA];
  __shared__ _Float16 Bs[128 * LDB];

  const int t    = threadIdx.x;
  const int lane = t & 31, wave = t >> 5;
  const int lg   = lane >> 4, lr = lane & 15;
  const int wm   = wave >> 2, wn = wave & 3;
  const int mBase = blockIdx.y * 128;
  const int nBase = blockIdx.x * 128;

  v8f acc[4][2];
#pragma unroll
  for (int i = 0; i < 4; ++i)
#pragma unroll
    for (int j = 0; j < 2; ++j) acc[i][j] = (v8f){0,0,0,0,0,0,0,0};

  const int arow = t >> 1, aseg = t & 1;
  const int bkp  = t >> 4, bns  = t & 15;

  for (int kb = 0; kb < EMBED; kb += 32) {
    __syncthreads();
    {   // A tile: async DMA global(f16) -> LDS, 2 x 16B per thread
      const _Float16* ap = A + (size_t)(mBase + arow) * EMBED + kb + aseg * 16;
      unsigned           l0 = (unsigned)(uintptr_t)&As[arow * LDA + aseg * 16];
      unsigned long long g0 = (unsigned long long)(uintptr_t)ap;
      asm volatile("global_load_async_to_lds_b128 %0, %1, off"
                   :: "v"(l0), "v"(g0) : "memory");
      asm volatile("global_load_async_to_lds_b128 %0, %1, off"
                   :: "v"(l0 + 16u), "v"(g0 + 16ull) : "memory");
    }
    {   // B tile: w_proj f32 -> f16 transposed, packed b32 stores
      const float* w0 = w + (size_t)(kb + 2 * bkp) * EMBED + nBase + bns * 8;
      const float* w1 = w0 + EMBED;
      v4f r00 = *(const v4f*)(w0);
      v4f r01 = *(const v4f*)(w0 + 4);
      v4f r10 = *(const v4f*)(w1);
      v4f r11 = *(const v4f*)(w1 + 4);
      float f0[8], f1[8];
#pragma unroll
      for (int i = 0; i < 4; ++i) {
        f0[i] = r00[i]; f0[i + 4] = r01[i];
        f1[i] = r10[i]; f1[i + 4] = r11[i];
      }
#pragma unroll
      for (int i = 0; i < 8; ++i) {
        union { _Float16 h[2]; unsigned u; } pk;
        pk.h[0] = (_Float16)f0[i];
        pk.h[1] = (_Float16)f1[i];
        *(unsigned*)&Bs[(bns * 8 + i) * LDB + 2 * bkp] = pk.u;
      }
    }
    asm volatile("s_wait_asynccnt 0" ::: "memory");
    __syncthreads();

    v16h af[4], bf[2];
#pragma unroll
    for (int mt = 0; mt < 4; ++mt) {
      const _Float16* ap = &As[(wm * 64 + mt * 16 + lr) * LDA + lg * 8];
      af[mt] = cat8(*(const v8h*)ap, *(const v8h*)(ap + 16));
    }
#pragma unroll
    for (int nt = 0; nt < 2; ++nt) {
      const _Float16* bp = &Bs[(wn * 32 + nt * 16 + lr) * LDB + lg * 8];
      bf[nt] = cat8(*(const v8h*)bp, *(const v8h*)(bp + 16));
    }
#pragma unroll
    for (int mt = 0; mt < 4; ++mt)
#pragma unroll
      for (int nt = 0; nt < 2; ++nt)
        acc[mt][nt] = wmma_f16(af[mt], bf[nt], acc[mt][nt]);
  }

#pragma unroll
  for (int nt = 0; nt < 2; ++nt) {
    const int col = nBase + wn * 32 + nt * 16 + lr;
    const float bv = bias[col];
#pragma unroll
    for (int mt = 0; mt < 4; ++mt) {
#pragma unroll
      for (int j = 0; j < 8; ++j) {
        const int row = mBase + wm * 64 + mt * 16 + j + 8 * lg;
        out[(size_t)row * EMBED + col] = acc[mt][nt][j] + bv;
      }
    }
  }
}

// ---------------------------------------------------------------------------
extern "C" void kernel_launch(void* const* d_in, const int* in_sizes, int n_in,
                              void* d_out, int out_size, void* d_ws, size_t ws_size,
                              hipStream_t stream) {
  (void)in_sizes; (void)n_in; (void)out_size; (void)ws_size;
  const float* x      = (const float*)d_in[0];
  const float* w_qkv  = (const float*)d_in[1];
  const float* b_qkv  = (const float*)d_in[2];
  const float* w_proj = (const float*)d_in[3];
  const float* b_proj = (const float*)d_in[4];
  float* out = (float*)d_out;

  const size_t qElems = (size_t)BATCH * NH * SEQ * HD;   // 12.58M f16 each
  _Float16* Q  = (_Float16*)d_ws;
  _Float16* K  = Q + qElems;
  _Float16* V  = K + qElems;
  _Float16* AO = V + qElems;                             // [B, N, E] f16

  const int M = BATCH * SEQ;                             // 16384
  qkv_gemm_kernel<<<dim3((3 * EMBED) / 128, M / 128), 256, 0, stream>>>(
      x, w_qkv, b_qkv, Q, K, V);
  attn_kernel<<<dim3(SEQ / 128, NH, BATCH), 256, 0, stream>>>(Q, K, V, AO);
  proj_gemm_kernel<<<dim3(EMBED / 128, M / 128), 256, 0, stream>>>(
      AO, w_proj, b_proj, out);
}
